// SenriAttention_38929583571556
// MI455X (gfx1250) — matrix operand order repaired and checked by reading, not compiled, not run on hardware
//
#include <hip/hip_runtime.h>
#include <hip/hip_bf16.h>
#include <math.h>

// ---------------------------------------------------------------------------
// SenriAttention for MI455X (gfx1250): bf16 WMMA everywhere, f32 accumulate.
// B=2, S=2048, H=D=2048, M=4 memories, 16 heads x 128 head-dim.
// Data movement: async global->LDS (ASYNCcnt) when available, double-buffered
// LDS tiles, global_prefetch hints one tile ahead.
// ---------------------------------------------------------------------------

typedef __attribute__((ext_vector_type(8)))  __bf16 v8bf;
typedef __attribute__((ext_vector_type(16))) __bf16 v16bf;
typedef __attribute__((ext_vector_type(8)))  float  v8f;
typedef int v4i __attribute__((vector_size(16)));   // matches async-LDS builtin

static constexpr int kB  = 2;
static constexpr int kS  = 2048;
static constexpr int kH  = 2048;
static constexpr int kD  = 2048;
static constexpr int kM  = 4;
static constexpr int kNH = 16;
static constexpr int kHD = 128;          // head dim
static constexpr int kBS = kB * kS;      // 4096 rows

#define WMMA_BF16(A_, B_, C_) \
  __builtin_amdgcn_wmma_f32_16x16x32_bf16(false, (A_), false, (B_), (short)0, (C_), false, false)

// --- CDNA5 async global->LDS path (guarded: falls back to reg staging) -----
#if defined(__AMDGCN__) && __has_builtin(__builtin_amdgcn_global_load_async_to_lds_b128)
#define ASYNC_LDS 1
#else
#define ASYNC_LDS 0
#endif

__device__ __forceinline__ void wait_async0() {
#if ASYNC_LDS
#if __has_builtin(__builtin_amdgcn_s_wait_asynccnt)
  __builtin_amdgcn_s_wait_asynccnt(0);
#else
  asm volatile("s_wait_asynccnt 0x0" ::: "memory");
#endif
#endif
}

// Copy 16 contiguous bf16 (32 bytes) global -> LDS.
__device__ __forceinline__ void cp16(__bf16* dst, const __bf16* src) {
#if ASYNC_LDS
  __builtin_amdgcn_global_load_async_to_lds_b128((v4i*)src, (v4i*)dst, 0, 0);
  __builtin_amdgcn_global_load_async_to_lds_b128((v4i*)(src + 8), (v4i*)(dst + 8), 0, 0);
#else
  *(v8bf*)dst       = *(const v8bf*)src;
  *(v8bf*)(dst + 8) = *(const v8bf*)(src + 8);
#endif
}

__device__ __forceinline__ v16bf cat16(v8bf lo, v8bf hi) {
  return __builtin_shufflevector(lo, hi, 0,1,2,3,4,5,6,7,8,9,10,11,12,13,14,15);
}

// A-matrix 16x32 bf16 fragment (ISA 7.12.2): lane half picks K-subblock.
//   lanes 0-15 : h[0..7]=K[0..7],  h[8..15]=K[16..23]
//   lanes16-31 : h[0..7]=K[8..15], h[8..15]=K[24..31]
__device__ __forceinline__ v16bf frag_a(const __bf16* rowp, int kb8) {
  v8bf lo = *(const v8bf*)(rowp + kb8);
  v8bf hi = *(const v8bf*)(rowp + kb8 + 16);
  return cat16(lo, hi);
}

// B-matrix 32x16 bf16 fragment: lane holds column n=lane&15, K values
// [kb8..kb8+7] and [kb8+16..kb8+23], kb8 = 8*(lane>>4) (same as A).
__device__ __forceinline__ v16bf frag_b(const __bf16* p) {
  v8bf lo = *(const v8bf*)(p);
  v8bf hi = *(const v8bf*)(p + 16);
  return cat16(lo, hi);
}

__device__ __forceinline__ v8f zero8() {
  v8f z = {0.f, 0.f, 0.f, 0.f, 0.f, 0.f, 0.f, 0.f};
  return z;
}

// ---------------------------------------------------------------------------
// Elementwise helpers
// ---------------------------------------------------------------------------
__global__ void cvt_bf16_kernel(const float* __restrict__ in,
                                __bf16* __restrict__ out, int n) {
  int i = blockIdx.x * blockDim.x + threadIdx.x;
  int stride = gridDim.x * blockDim.x;
  for (; i < n; i += stride) out[i] = (__bf16)in[i];
}

__global__ void zero_kernel(float* __restrict__ p, int n) {
  int i = blockIdx.x * blockDim.x + threadIdx.x;
  int stride = gridDim.x * blockDim.x;
  for (; i < n; i += stride) p[i] = 0.f;
}

// q -> q_bf (pre-scaled by 1/sqrt(hd)) and sigma_q = elu(q)+1 -> bf16
__global__ void qpoint_kernel(const float* __restrict__ q,
                              __bf16* __restrict__ qb,
                              __bf16* __restrict__ sq, int n) {
  int i = blockIdx.x * blockDim.x + threadIdx.x;
  int stride = gridDim.x * blockDim.x;
  const float scale = 0.08838834764831845f;  // 128^-0.5
  for (; i < n; i += stride) {
    float v = q[i];
    qb[i] = (__bf16)(v * scale);
    sq[i] = (__bf16)(v > 0.f ? v + 1.f : __expf(v));  // elu(v)+1
  }
}

// raw[m, i] = dot(sigma_q[i, :], memory_norms[m, :]) — one wave per (m,i)
__global__ __launch_bounds__(256)
void raw_kernel(const __bf16* __restrict__ sq,
                const float* __restrict__ norms,
                float* __restrict__ raw) {
  int gw = blockIdx.x * 8 + (threadIdx.x >> 5);   // 0 .. M*BS-1
  int lane = threadIdx.x & 31;
  int m = gw / kBS, i = gw % kBS;
  const __bf16* sp = sq + (size_t)i * kD;
  const float* np = norms + (size_t)m * kD;
  float acc = 0.f;
  for (int d = lane; d < kD; d += 32) acc += (float)sp[d] * np[d];
  #pragma unroll
  for (int o = 16; o > 0; o >>= 1) acc += __shfl_xor(acc, o, 32);
  if (lane == 0) raw[gw] = acc;
}

// rel[m,b] = mean_s raw; wgt[m,b] = sigmoid(gate) * softmax_m(rel)
__global__ __launch_bounds__(256)
void relwgt_kernel(const float* __restrict__ raw,
                   const float* __restrict__ gate,
                   float* __restrict__ wgt) {
  __shared__ float part[256];
  __shared__ float rel[kM * kB];
  int t = threadIdx.x;
  for (int mb = 0; mb < kM * kB; ++mb) {
    int m = mb / kB, b = mb % kB;
    float loc = 0.f;
    for (int s = t; s < kS; s += 256) loc += raw[(size_t)m * kBS + b * kS + s];
    part[t] = loc;
    __syncthreads();
    for (int o = 128; o > 0; o >>= 1) {
      if (t < o) part[t] += part[t + o];
      __syncthreads();
    }
    if (t == 0) rel[mb] = part[0] / (float)kS;
    __syncthreads();
  }
  if (t < kB) {
    int b = t;
    float mx = -3.0e38f;
    for (int m = 0; m < kM; ++m) mx = fmaxf(mx, rel[m * kB + b]);
    float e[kM], den = 0.f;
    for (int m = 0; m < kM; ++m) { e[m] = __expf(rel[m * kB + b] - mx); den += e[m]; }
    float g = 1.f / (1.f + __expf(-gate[0]));
    for (int m = 0; m < kM; ++m) wgt[m * kB + b] = g * e[m] / den;
  }
}

// alpha[m,i] = wgt[m, b(i)] / max(raw[m,i], eps)
__global__ void alpha_kernel(const float* __restrict__ raw,
                             const float* __restrict__ wgt,
                             float* __restrict__ alpha) {
  int idx = blockIdx.x * blockDim.x + threadIdx.x;
  int stride = gridDim.x * blockDim.x;
  int n = kM * kBS;
  for (; idx < n; idx += stride) {
    int m = idx >> 12;          // / 4096
    int i = idx & (kBS - 1);
    int b = i >> 11;            // / 2048
    alpha[idx] = wgt[m * kB + b] / fmaxf(raw[idx], 1e-6f);
  }
}

// ---------------------------------------------------------------------------
// GEMM (NT): C[Mr,N] = A[Mr,K] * B[N,K]^T, A/B bf16 row-major (contraction on
// the fast dim of both). mode 0 -> f32 out (Cf), mode 1 -> bf16 out (Cb).
// Block tile 128x128, K-step 32, 8 waves (4x2), 8 WMMA per wave per step.
// Double-buffered LDS; next tile staged (async if available) before compute;
// single barrier per K-step; prefetch one tile further ahead into GL2.
// ---------------------------------------------------------------------------
__global__ __launch_bounds__(256)
void gemm_nt_kernel(const __bf16* __restrict__ A, const __bf16* __restrict__ Bm,
                    float* __restrict__ Cf, __bf16* __restrict__ Cb,
                    int Mr, int N, int K, int mode) {
  __shared__ __bf16 lA[2][128][40];   // 32 K-halves + pad 8 (conflict-free b128)
  __shared__ __bf16 lB[2][128][40];
  (void)Mr;
  const int t = threadIdx.x;
  const int lane = t & 31, w = t >> 5;
  const int wm = w >> 1, wn = w & 1;
  const int l15 = lane & 15, lhi = lane >> 4;
  const int kb8 = 8 * lhi, kb16 = 16 * lhi;
  const int m0 = blockIdx.y * 128, n0 = blockIdx.x * 128;

  v8f acc[2][4];
  #pragma unroll
  for (int mi = 0; mi < 2; ++mi)
    #pragma unroll
    for (int ni = 0; ni < 4; ++ni) acc[mi][ni] = zero8();

  const int srow = t >> 1, shalf = (t & 1) * 16;
  const __bf16* gA = A + (size_t)(m0 + srow) * K + shalf;
  const __bf16* gB = Bm + (size_t)(n0 + srow) * K + shalf;

  // prologue: stage first tile into buffer 0
  cp16(&lA[0][srow][shalf], gA);
  cp16(&lB[0][srow][shalf], gB);
  wait_async0();
  __syncthreads();

  for (int kk = 0; kk < K; kk += 32) {
    const int cur = (kk >> 5) & 1;
    if (kk + 32 < K) {                       // issue next tile before compute
      cp16(&lA[cur ^ 1][srow][shalf], gA + kk + 32);
      cp16(&lB[cur ^ 1][srow][shalf], gB + kk + 32);
    }
    if (kk + 64 < K) {                       // GL2 prefetch one tile further
      __builtin_prefetch(gA + kk + 64, 0, 1);
      __builtin_prefetch(gB + kk + 64, 0, 1);
    }

    v16bf af[2], bfr[4];
    #pragma unroll
    for (int mi = 0; mi < 2; ++mi)
      af[mi] = frag_a(&lA[cur][wm * 32 + mi * 16 + l15][0], kb8);
    #pragma unroll
    for (int ni = 0; ni < 4; ++ni)
      bfr[ni] = frag_b(&lB[cur][wn * 64 + ni * 16 + l15][kb8]);
    #pragma unroll
    for (int mi = 0; mi < 2; ++mi)
      #pragma unroll
      for (int ni = 0; ni < 4; ++ni)
        acc[mi][ni] = WMMA_BF16(af[mi], bfr[ni], acc[mi][ni]);

    wait_async0();
    __syncthreads();
  }

  #pragma unroll
  for (int mi = 0; mi < 2; ++mi)
    #pragma unroll
    for (int ni = 0; ni < 4; ++ni)
      #pragma unroll
      for (int r = 0; r < 8; ++r) {
        int grow = m0 + wm * 32 + mi * 16 + 8 * lhi + r;
        int gcol = n0 + wn * 64 + ni * 16 + l15;
        size_t idx = (size_t)grow * N + gcol;
        if (mode == 0) Cf[idx] = acc[mi][ni][r];
        else           Cb[idx] = (__bf16)acc[mi][ni][r];
      }
}

// ---------------------------------------------------------------------------
// GEMM (NN) with per-row scale accumulate:
//   C[i,n] += rowscale[i] * sum_k A[i,k]*B[k,n],  B row-major [K,N].
// B tile is transposed into LDS during staging (element ds_stores); A tile
// staged via (async) cp16. Prefetch hints one K-step ahead.
// ---------------------------------------------------------------------------
__global__ __launch_bounds__(256)
void gemm_nn_rowscale_kernel(const __bf16* __restrict__ A,
                             const __bf16* __restrict__ Bm,
                             const float* __restrict__ rowscale,
                             float* __restrict__ Cf,
                             int Mr, int N, int K) {
  __shared__ __bf16 lA[128][40];
  __shared__ __bf16 lB[128][40];   // transposed: [n_local][k]
  (void)Mr;
  const int t = threadIdx.x;
  const int lane = t & 31, w = t >> 5;
  const int wm = w >> 1, wn = w & 1;
  const int l15 = lane & 15, lhi = lane >> 4;
  const int kb8 = 8 * lhi, kb16 = 16 * lhi;
  const int m0 = blockIdx.y * 128, n0 = blockIdx.x * 128;

  v8f acc[2][4];
  #pragma unroll
  for (int mi = 0; mi < 2; ++mi)
    #pragma unroll
    for (int ni = 0; ni < 4; ++ni) acc[mi][ni] = zero8();

  const int srow = t >> 1, shalf = (t & 1) * 16;
  const int bk = t & 31, bseg = t >> 5;  // B staging: k-row + 16-col segment
  const __bf16* gA = A + (size_t)(m0 + srow) * K + shalf;

  for (int kk = 0; kk < K; kk += 32) {
    cp16(&lA[srow][shalf], gA + kk);
    const __bf16* gb = Bm + (size_t)(kk + bk) * N + n0 + bseg * 16;
    v8bf b0 = *(const v8bf*)gb;
    v8bf b1 = *(const v8bf*)(gb + 8);
    #pragma unroll
    for (int e = 0; e < 8; ++e) {
      lB[bseg * 16 + e][bk]     = b0[e];
      lB[bseg * 16 + 8 + e][bk] = b1[e];
    }
    if (kk + 32 < K) {
      __builtin_prefetch(gA + kk + 32, 0, 1);
      __builtin_prefetch(Bm + (size_t)(kk + 32 + bk) * N + n0 + bseg * 16, 0, 1);
    }
    wait_async0();
    __syncthreads();

    v16bf af[2], bfr[4];
    #pragma unroll
    for (int mi = 0; mi < 2; ++mi)
      af[mi] = frag_a(&lA[wm * 32 + mi * 16 + l15][0], kb8);
    #pragma unroll
    for (int ni = 0; ni < 4; ++ni)
      bfr[ni] = frag_b(&lB[wn * 64 + ni * 16 + l15][kb8]);
    #pragma unroll
    for (int mi = 0; mi < 2; ++mi)
      #pragma unroll
      for (int ni = 0; ni < 4; ++ni)
        acc[mi][ni] = WMMA_BF16(af[mi], bfr[ni], acc[mi][ni]);
    __syncthreads();
  }

  #pragma unroll
  for (int mi = 0; mi < 2; ++mi)
    #pragma unroll
    for (int ni = 0; ni < 4; ++ni)
      #pragma unroll
      for (int r = 0; r < 8; ++r) {
        int grow = m0 + wm * 32 + mi * 16 + 8 * lhi + r;
        int gcol = n0 + wn * 64 + ni * 16 + l15;
        size_t idx = (size_t)grow * N + gcol;
        Cf[idx] += rowscale[grow] * acc[mi][ni][r];
      }
}

// ---------------------------------------------------------------------------
// Causal attention: one workgroup (4 waves, 128 threads) per 16 q-rows per
// (batch, head). Full causal score band (16 x ncols f32) kept in LDS —
// leverages the 320KB/WGP LDS of CDNA5. Adds (1-g)*out into comb.
// ---------------------------------------------------------------------------
__global__ __launch_bounds__(128)
void attn_kernel(const __bf16* __restrict__ qb, const __bf16* __restrict__ kb,
                 const __bf16* __restrict__ vb, const float* __restrict__ gate,
                 float* __restrict__ comb) {
  extern __shared__ char smem[];
  float (*sc)[kS]      = (float (*)[kS])smem;                          // 16 x 2048 f32
  __bf16 (*qt)[136]    = (__bf16 (*)[136])(smem + 16 * kS * 4);        // 16 x 128 (+pad)
  __bf16 (*vt)[40]     = (__bf16 (*)[40])(smem + 16 * kS * 4 + 16 * 136 * 2); // 128 x 32 (+pad)

  const int t = threadIdx.x, lane = t & 31, w = t >> 5;
  const int l15 = lane & 15, lhi = lane >> 4;
  const int kb8 = 8 * lhi, kb16 = 16 * lhi;
  const int qi = blockIdx.x, bh = blockIdx.y;
  const int b = bh / kNH, h = bh % kNH;
  const int q0 = qi * 16;
  const int ncols = q0 + 16;     // causal column count (multiple of 16)

  // --- stage q tile: 16 rows x 128 head-dims (async if available)
  {
    int row = t >> 3, seg = t & 7;
    const __bf16* gq = qb + (size_t)(b * kS + q0 + row) * kD + h * kHD + seg * 16;
    cp16(&qt[row][seg * 16], gq);
  }
  wait_async0();
  __syncthreads();

  // q fragments: invariant across k-tiles (4 chunks of K=32 over hd=128)
  v16bf qf[4];
  #pragma unroll
  for (int c = 0; c < 4; ++c) qf[c] = frag_a(&qt[l15][c * 32], kb8);

  // --- phase A: scores = (q*scale) @ k^T, causal mask, into LDS
  const int nkt = ncols >> 4;
  for (int kt = w; kt < nkt; kt += 4) {
    const int key = kt * 16 + l15;  // this lane's key column
    const __bf16* gk = kb + (size_t)(b * kS + key) * kD + h * kHD;
    if (kt + 4 < nkt)  // prefetch this lane's next key row into GL2
      __builtin_prefetch(gk + (size_t)64 * kD, 0, 1);
    v8f s = zero8();
    #pragma unroll
    for (int c = 0; c < 4; ++c) {
      v16bf kf = frag_b(gk + c * 32 + kb8);  // contiguous along hd
      s = WMMA_BF16(qf[c], kf, s);
    }
    #pragma unroll
    for (int r = 0; r < 8; ++r) {
      int rowm = 8 * lhi + r;
      sc[rowm][key] = (key <= q0 + rowm) ? s[r] : -1e9f;
    }
  }
  __syncthreads();

  // --- phase B: row softmax over ncols (4 rows/wave, 8 lanes/row)
  {
    int row = w * 4 + (lane >> 3);
    int lg = lane & 7;
    float mx = -3.0e38f;
    for (int c = lg; c < ncols; c += 8) mx = fmaxf(mx, sc[row][c]);
    mx = fmaxf(mx, __shfl_xor(mx, 1, 32));
    mx = fmaxf(mx, __shfl_xor(mx, 2, 32));
    mx = fmaxf(mx, __shfl_xor(mx, 4, 32));
    float sum = 0.f;
    for (int c = lg; c < ncols; c += 8) sum += __expf(sc[row][c] - mx);
    sum += __shfl_xor(sum, 1, 32);
    sum += __shfl_xor(sum, 2, 32);
    sum += __shfl_xor(sum, 4, 32);
    float inv = 1.f / sum;
    for (int c = lg; c < ncols; c += 8) sc[row][c] = __expf(sc[row][c] - mx) * inv;
  }

  // --- phase C: out = P @ V ; each wave owns a 32-wide head-dim slice
  v8f o[2] = {zero8(), zero8()};
  const int n0 = w * 32;
  for (int kc = 0; kc < ncols; kc += 32) {
    __syncthreads();  // previous vt consumers done (also covers phase-B stores)
    {  // stage V^T tile: vt[n][j] = V[kc+j][h*128+n]
      int j = t >> 2, seg = t & 3;
      const __bf16* gv = vb + (size_t)(b * kS + kc + j) * kD + h * kHD + seg * 32;
      v8bf x0 = *(const v8bf*)(gv);
      v8bf x1 = *(const v8bf*)(gv + 8);
      v8bf x2 = *(const v8bf*)(gv + 16);
      v8bf x3 = *(const v8bf*)(gv + 24);
      if (kc + 32 < ncols)
        __builtin_prefetch(gv + (size_t)32 * kD, 0, 1);
      #pragma unroll
      for (int e = 0; e < 8; ++e) {
        vt[seg * 32 + e][j]      = x0[e];
        vt[seg * 32 + 8 + e][j]  = x1[e];
        vt[seg * 32 + 16 + e][j] = x2[e];
        vt[seg * 32 + 24 + e][j] = x3[e];
      }
    }
    __syncthreads();

    // P fragment (A-matrix layout) from LDS f32 probs, cvt -> bf16
    const float* srow = sc[l15];
    int c0 = kc + kb8;
    v8bf plo, phi;
    {
      float4 f0 = *(const float4*)(srow + c0);
      float4 f1 = *(const float4*)(srow + c0 + 4);
      plo[0] = (__bf16)f0.x; plo[1] = (__bf16)f0.y; plo[2] = (__bf16)f0.z; plo[3] = (__bf16)f0.w;
      plo[4] = (__bf16)f1.x; plo[5] = (__bf16)f1.y; plo[6] = (__bf16)f1.z; plo[7] = (__bf16)f1.w;
    }
    if (kc + 16 < ncols) {
      float4 f0 = *(const float4*)(srow + c0 + 16);
      float4 f1 = *(const float4*)(srow + c0 + 20);
      phi[0] = (__bf16)f0.x; phi[1] = (__bf16)f0.y; phi[2] = (__bf16)f0.z; phi[3] = (__bf16)f0.w;
      phi[4] = (__bf16)f1.x; phi[5] = (__bf16)f1.y; phi[6] = (__bf16)f1.z; phi[7] = (__bf16)f1.w;
    } else {
      #pragma unroll
      for (int e = 0; e < 8; ++e) phi[e] = (__bf16)0.0f;
    }
    v16bf pf = cat16(plo, phi);

    #pragma unroll
    for (int nt = 0; nt < 2; ++nt) {
      v16bf vf = frag_b(&vt[n0 + nt * 16 + l15][kb8]);
      o[nt] = WMMA_BF16(pf, vf, o[nt]);
    }
  }

  // --- epilogue: comb += (1 - sigmoid(gate)) * out
  float gi = 1.f - 1.f / (1.f + __expf(-gate[0]));
  #pragma unroll
  for (int nt = 0; nt < 2; ++nt)
    #pragma unroll
    for (int r = 0; r < 8; ++r) {
      int rowm = 8 * lhi + r;
      int gcol = h * kHD + n0 + nt * 16 + l15;
      size_t idx = (size_t)(b * kS + q0 + rowm) * kD + gcol;
      comb[idx] += gi * o[nt][r];
    }
}

// ---------------------------------------------------------------------------
// Host orchestration
// ---------------------------------------------------------------------------
extern "C" void kernel_launch(void* const* d_in, const int* in_sizes, int n_in,
                              void* d_out, int out_size, void* d_ws, size_t ws_size,
                              hipStream_t stream) {
  (void)in_sizes; (void)n_in; (void)out_size; (void)ws_size;
  const float* x     = (const float*)d_in[0];
  const float* w_q   = (const float*)d_in[1];
  const float* w_k   = (const float*)d_in[2];
  const float* w_v   = (const float*)d_in[3];
  const float* w_o   = (const float*)d_in[4];
  const float* gate  = (const float*)d_in[5];
  const float* mems  = (const float*)d_in[6];
  const float* norms = (const float*)d_in[7];
  float* y = (float*)d_out;

  char* ws = (char*)d_ws;
  size_t off = 0;
  auto take = [&](size_t bytes) -> char* {
    char* p = ws + off;
    off += (bytes + 255) & ~(size_t)255;
    return p;
  };
  __bf16* x_bf   = (__bf16*)take((size_t)kBS * kH * 2);
  __bf16* wq_bf  = (__bf16*)take((size_t)kD * kH * 2);
  __bf16* wk_bf  = (__bf16*)take((size_t)kD * kH * 2);
  __bf16* wv_bf  = (__bf16*)take((size_t)kD * kH * 2);
  __bf16* wo_bf  = (__bf16*)take((size_t)kH * kD * 2);
  __bf16* mem_bf = (__bf16*)take((size_t)kM * kD * kD * 2);
  float*  q_f    = (float*)take((size_t)kBS * kD * 4);
  __bf16* q_bf   = (__bf16*)take((size_t)kBS * kD * 2);
  __bf16* sq_bf  = (__bf16*)take((size_t)kBS * kD * 2);
  __bf16* k_bf   = (__bf16*)take((size_t)kBS * kD * 2);
  __bf16* v_bf   = (__bf16*)take((size_t)kBS * kD * 2);
  float*  comb   = (float*)take((size_t)kBS * kD * 4);
  __bf16* cb_bf  = (__bf16*)take((size_t)kBS * kD * 2);
  float*  raw    = (float*)take((size_t)kM * kBS * 4);
  float*  alpha  = (float*)take((size_t)kM * kBS * 4);
  float*  wgt    = (float*)take((size_t)kM * kB * 4);

  auto cvt = [&](const float* src, __bf16* dst, int n) {
    int blocks = (n + 255) / 256;
    if (blocks > 16384) blocks = 16384;
    cvt_bf16_kernel<<<blocks, 256, 0, stream>>>(src, dst, n);
  };

  // 1) convert everything matmul-shaped to bf16
  cvt(x, x_bf, kBS * kH);
  cvt(w_q, wq_bf, kD * kH);
  cvt(w_k, wk_bf, kD * kH);
  cvt(w_v, wv_bf, kD * kH);
  cvt(w_o, wo_bf, kH * kD);
  cvt(mems, mem_bf, kM * kD * kD);

  // 2) projections: q (f32 out), k/v (bf16 out)
  dim3 ggrid(kD / 128, kBS / 128);
  gemm_nt_kernel<<<ggrid, 256, 0, stream>>>(x_bf, wq_bf, q_f, (__bf16*)nullptr,
                                            kBS, kD, kH, 0);
  gemm_nt_kernel<<<ggrid, 256, 0, stream>>>(x_bf, wk_bf, (float*)nullptr, k_bf,
                                            kBS, kD, kH, 1);
  gemm_nt_kernel<<<ggrid, 256, 0, stream>>>(x_bf, wv_bf, (float*)nullptr, v_bf,
                                            kBS, kD, kH, 1);

  // 3) q pointwise: scaled q_bf + sigma_q bf16
  qpoint_kernel<<<16384, 256, 0, stream>>>(q_f, q_bf, sq_bf, kBS * kD);

  // 4) retrieval normalizers: raw, relevance softmax, row scales
  raw_kernel<<<(kM * kBS) / 8, 256, 0, stream>>>(sq_bf, norms, raw);
  relwgt_kernel<<<1, 256, 0, stream>>>(raw, gate, wgt);
  alpha_kernel<<<64, 256, 0, stream>>>(raw, wgt, alpha);

  // 5) comb = sum_m alpha[m,i] * (sigma_q @ mem[m])   (gate folded into alpha)
  zero_kernel<<<8192, 256, 0, stream>>>(comb, kBS * kD);
  for (int m = 0; m < kM; ++m) {
    gemm_nn_rowscale_kernel<<<ggrid, 256, 0, stream>>>(
        sq_bf, mem_bf + (size_t)m * kD * kD, alpha + (size_t)m * kBS, comb,
        kBS, kD, kD);
  }

  // 6) comb += (1-g) * causal attention
  const int ATTN_LDS = 16 * kS * 4 + 16 * 136 * 2 + 128 * 40 * 2;  // 145664 B
  (void)hipFuncSetAttribute((const void*)attn_kernel,
                            hipFuncAttributeMaxDynamicSharedMemorySize, ATTN_LDS);
  attn_kernel<<<dim3(kS / 16, kB * kNH), 128, ATTN_LDS, stream>>>(
      q_bf, k_bf, v_bf, gate, comb);

  // 7) y = comb @ w_o^T
  cvt(comb, cb_bf, kBS * kD);
  dim3 fgrid(kH / 128, kBS / 128);
  gemm_nt_kernel<<<fgrid, 256, 0, stream>>>(cb_bf, wo_bf, y, (__bf16*)nullptr,
                                            kBS, kH, kD, 0);
}
